// DySPN_Modulev2_40338332844082
// MI455X (gfx1250) — matrix operand order repaired and checked by reading, not compiled, not run on hardware
//
#include <hip/hip_runtime.h>
#include <math.h>

// ---------------- problem constants ----------------
#define ITER_ 3
#define NUM_  8
#define CH_   24           // guide channels
#define B_    2
#define H_    352
#define W_    1216
#define KR_   216          // real reduction = CH_*9
#define KP_   224          // padded reduction = 7 * 32
#define NKK_  7            // K-steps of 32
#define MT_   5            // M tiles of 16 (80 padded channels; 72 real)
#define TILE_PX 128        // pixels per workgroup (8 waves x 16)
#define TILES_X ((W_ + TILE_PX - 1) / TILE_PX)   // 10

// ---------------- WMMA / packing types (gfx1250, wave32) ----------------
typedef __bf16    v16bf __attribute__((ext_vector_type(16)));
typedef float     v8f   __attribute__((ext_vector_type(8)));
typedef unsigned  v4u   __attribute__((ext_vector_type(4)));
typedef _Float16  h2    __attribute__((ext_vector_type(2)));

union Frag { v16bf bf; v4u u[2]; };
union H2U  { h2 h; unsigned u; };

__device__ __forceinline__ unsigned short f2bf(float f) {
  // round-to-nearest-even f32 -> bf16
  unsigned u = __float_as_uint(f);
  unsigned r = 0x7FFFu + ((u >> 16) & 1u);
  return (unsigned short)((u + r) >> 16);
}

// =====================================================================
// Kernel 1: pre-swizzle conv weights into exact WMMA A-fragment order.
// Aprep chunk c = (kk*5 + mt)*32 + lane holds the 16 bf16 that lane feeds
// to the v_wmma A operand (16-bit A 16x32 layout: lanes<16 -> K{0..7,
// 16..23}, lanes>=16 -> +8). Also fuses the two bias vectors into bias[80].
// =====================================================================
__global__ void dyspn_prep(const float* __restrict__ w_off,
                           const float* __restrict__ b_off,
                           const float* __restrict__ w_aff,
                           const float* __restrict__ b_aff,
                           unsigned short* __restrict__ Aprep,
                           float* __restrict__ bias) {
  int tid = blockIdx.x * blockDim.x + threadIdx.x;
  int nth = gridDim.x * blockDim.x;
  const int TOT = NKK_ * MT_ * 32 * 16;
  for (int i = tid; i < TOT; i += nth) {
    int s    = i & 15;            // bf16 slot within lane
    int lane = (i >> 4) & 31;
    int mt   = (i >> 9) % MT_;
    int kk   = i / (MT_ * 32 * 16);
    int m = mt * 16 + (lane & 15);
    int k = kk * 32 + ((lane < 16) ? 0 : 8) + ((s < 8) ? s : (s + 8));
    float v = 0.f;
    if (m < 72 && k < KR_)
      v = (m < 48) ? w_off[m * KR_ + k] : w_aff[(m - 48) * KR_ + k];
    Aprep[i] = f2bf(v);
  }
  for (int c = tid; c < 80; c += nth)
    bias[c] = (c < 48) ? b_off[c] : ((c < 72) ? b_aff[c - 48] : 0.f);
}

// per-lane softmax over 8 accumulator components -> f16 planes
__device__ __forceinline__ void softmax8_store(v8f a, _Float16* __restrict__ dst,
                                               size_t pix) {
  const size_t HW = (size_t)H_ * W_;
  float m = a[0];
#pragma unroll
  for (int j = 1; j < 8; ++j) m = fmaxf(m, a[j]);
  float e[8]; float s = 0.f;
#pragma unroll
  for (int j = 0; j < 8; ++j) { e[j] = __expf(a[j] - m); s += e[j]; }
  float inv = 1.f / s;
#pragma unroll
  for (int j = 0; j < 8; ++j)
    dst[(size_t)j * HW + pix] = (_Float16)(e[j] * inv);
}

// =====================================================================
// Kernel 2: fused implicit-GEMM conv (bf16 WMMA) + offset/affinity epilogue.
// block = 256 threads (8 waves), one output row y, 128 consecutive pixels.
// The normalized-grid math reduces exactly to sample = offset + pixel, so
// we emit the raw offsets packed f16x2 (1 dword/neighbor) and f16 softmaxed
// affinities -- halving the HBM traffic that bounds this op on MI455X.
// =====================================================================
__global__ __launch_bounds__(256)
void dyspn_conv_wmma(const float* __restrict__ guide,
                     const unsigned short* __restrict__ Aprep,
                     const float* __restrict__ bias,
                     unsigned* __restrict__ offs,     // packed f16x2 planes
                     _Float16* __restrict__ aff) {    // f16 planes
  __shared__ __attribute__((aligned(16))) unsigned short Bl[TILE_PX * KP_];

  const int tid = threadIdx.x;
  const int x0  = blockIdx.x * TILE_PX;
  const int yg  = blockIdx.y;
  const int bz  = blockIdx.z;

  // ---- Phase 1: im2col tile -> LDS as bf16, K-contiguous per pixel ----
  for (int i = tid; i < TILE_PX * (KP_ / 2); i += 256) {
    int n  = i / (KP_ / 2);
    int kp = i - n * (KP_ / 2);
    int xg = x0 + n;
    unsigned pack = 0u;
#pragma unroll
    for (int h = 0; h < 2; ++h) {
      int k = kp * 2 + h;
      float v = 0.f;
      if (k < KR_) {
        int cin = k / 9;
        int tap = k - cin * 9;
        int dy  = tap / 3;
        int dx  = tap - dy * 3;
        int yy = yg + dy - 1;
        int xx = xg + dx - 1;
        if (yy >= 0 && yy < H_ && xx >= 0 && xx < W_)
          v = guide[((size_t)(bz * CH_ + cin) * H_ + yy) * W_ + xx];
      }
      pack |= (unsigned)f2bf(v) << (16 * h);
    }
    *(unsigned*)&Bl[(size_t)n * KP_ + kp * 2] = pack;
  }
  __syncthreads();

  // ---- Phase 2: 5 Mtiles x 7 Ksteps of v_wmma_f32_16x16x32_bf16 ----
  const int lane = tid & 31;
  const int wv   = tid >> 5;               // 8 waves -> 8 N tiles of 16 px
  v8f acc[MT_] = {};

  const int nrow = wv * 16 + (lane & 15);
  const int kbB  = (lane < 16) ? 0 : 8;    // B 32x16: lanes<16 hold K{0..7,16..23}, lanes>=16 +8
  const v4u* Ap  = (const v4u*)Aprep;

  for (int kk = 0; kk < NKK_; ++kk) {
    Frag bfr;
    const unsigned short* bp = &Bl[(size_t)nrow * KP_ + kk * 32 + kbB];
    bfr.u[0] = *(const v4u*)bp;            // ds_load_b128
    bfr.u[1] = *(const v4u*)(bp + 16);     // ds_load_b128
#pragma unroll
    for (int mt = 0; mt < MT_; ++mt) {
      Frag afr;
      size_t ci = ((size_t)(kk * MT_ + mt) * 32 + lane) * 2;
      afr.u[0] = Ap[ci];                   // coalesced global_load_b128
      afr.u[1] = Ap[ci + 1];
      acc[mt] = __builtin_amdgcn_wmma_f32_16x16x32_bf16(
          false, afr.bf, false, bfr.bf, (short)0, acc[mt], false, false);
    }
  }

  // ---- Phase 3: epilogue. lane holds pixel p, channels r + 8*hi mod 16 ----
  const int hi = lane >> 4;
  const int p  = lane & 15;
  const int xg = x0 + wv * 16 + p;
  if (xg < W_) {
#pragma unroll
    for (int mt = 0; mt < MT_; ++mt)
#pragma unroll
      for (int r = 0; r < 8; ++r)
        acc[mt][r] += bias[mt * 16 + r + 8 * hi];

    const size_t HW  = (size_t)H_ * W_;
    const size_t pix = (size_t)yg * W_ + xg;

    // offsets: channels 0..47 -> (iter=mt<3, n=j+4*hi, xy pair), packed f16x2
#pragma unroll
    for (int mt = 0; mt < 3; ++mt) {
#pragma unroll
      for (int j = 0; j < 4; ++j) {
        int n = hi * 4 + j;
        H2U t;
        t.h = h2{(_Float16)acc[mt][2 * j], (_Float16)acc[mt][2 * j + 1]};
        offs[((size_t)((bz * ITER_ + mt) * NUM_ + n)) * HW + pix] = t.u;
      }
    }
    // affinities: iter0 -> acc[3] (hi==0); iter1 -> acc[3] (hi==1);
    // iter2 -> acc[4] (hi==0). Softmax entirely within one lane.
    if (hi == 0) {
      softmax8_store(acc[3], aff + ((size_t)(bz * ITER_ + 0) * NUM_) * HW, pix);
      softmax8_store(acc[4], aff + ((size_t)(bz * ITER_ + 2) * NUM_) * HW, pix);
    } else {
      softmax8_store(acc[3], aff + ((size_t)(bz * ITER_ + 1) * NUM_) * HW, pix);
    }
  }
}

// =====================================================================
// Kernel 3: one propagation round (serial dependence across rounds).
// 8-neighbor bilinear gather with zeros padding + affinity blend + conf mix.
// Sample position rebuilt in fp32 as (f16 offset) + integer pixel coord.
// =====================================================================
__device__ __forceinline__ float tapv(const float* __restrict__ img,
                                      int xi, int yi) {
  if (xi < 0 || xi >= W_ || yi < 0 || yi >= H_) return 0.f;
  return img[(size_t)yi * W_ + xi];
}

__global__ __launch_bounds__(256)
void dyspn_iter(const float* __restrict__ xprev,
                const float* __restrict__ sp_dep,
                const float* __restrict__ conf_in,
                const unsigned* __restrict__ offs,
                const _Float16* __restrict__ aff,
                float* __restrict__ xnext, int iter) {
  const size_t HW = (size_t)H_ * W_;
  int idx = blockIdx.x * blockDim.x + threadIdx.x;
  if (idx >= B_ * H_ * W_) return;
  int b   = idx / (int)HW;
  int rem = idx - b * (int)HW;
  int py  = rem / W_;
  int px  = rem - py * W_;

  float sd  = sp_dep[idx];
  float sgn = (float)((sd > 0.f) - (sd < 0.f));
  float c   = sgn / (1.f + __expf(-conf_in[idx]));

  const float* img = xprev + (size_t)b * HW;
  float out = 0.f;
#pragma unroll
  for (int n = 0; n < NUM_; ++n) {
    size_t o = ((size_t)((b * ITER_ + iter) * NUM_ + n)) * HW + rem;
    H2U t; t.u = offs[o];
    float a  = (float)aff[o];
    float sx = (float)t.h[0] + (float)px;
    float sy = (float)t.h[1] + (float)py;
    float xf = floorf(sx), yf = floorf(sy);
    float wx = sx - xf,    wy = sy - yf;
    int xi = (int)xf, yi = (int)yf;
    float v00 = tapv(img, xi,     yi);
    float v01 = tapv(img, xi + 1, yi);
    float v10 = tapv(img, xi,     yi + 1);
    float v11 = tapv(img, xi + 1, yi + 1);
    float v = v00 * (1.f - wx) * (1.f - wy) + v01 * wx * (1.f - wy)
            + v10 * (1.f - wx) * wy         + v11 * wx * wy;
    out = fmaf(a, v, out);
  }
  xnext[idx] = (1.f - c) * out + c * sd;
}

// =====================================================================
// launch: prep -> fused WMMA conv -> 3 serial propagation rounds
// workspace needed ~130 MB (offs 82MB + aff 41MB + Aprep/bias + 2 ping-pong)
// =====================================================================
extern "C" void kernel_launch(void* const* d_in, const int* in_sizes, int n_in,
                              void* d_out, int out_size, void* d_ws,
                              size_t ws_size, hipStream_t stream) {
  (void)in_sizes; (void)n_in; (void)out_size; (void)ws_size;
  const float* input  = (const float*)d_in[0];
  const float* guide  = (const float*)d_in[1];
  const float* sp_dep = (const float*)d_in[2];
  const float* confid = (const float*)d_in[3];
  const float* w_off  = (const float*)d_in[4];
  const float* b_off  = (const float*)d_in[5];
  const float* w_aff  = (const float*)d_in[6];
  const float* b_aff  = (const float*)d_in[7];
  float* out = (float*)d_out;

  const size_t HW     = (size_t)H_ * W_;
  const size_t nPlane = (size_t)B_ * ITER_ * NUM_ * HW;   // 20,545,536

  char* ws = (char*)d_ws;
  unsigned* offs = (unsigned*)ws;            ws += nPlane * sizeof(unsigned);
  _Float16* aff  = (_Float16*)ws;            ws += nPlane * sizeof(_Float16);
  unsigned short* Aprep = (unsigned short*)ws; ws += (size_t)NKK_ * MT_ * 32 * 16 * 2;
  float* bias = (float*)ws;                  ws += 80 * sizeof(float);
  float* xb0  = (float*)ws;                  ws += (size_t)B_ * HW * sizeof(float);
  float* xb1  = (float*)ws;

  dyspn_prep<<<dim3(16), dim3(256), 0, stream>>>(w_off, b_off, w_aff, b_aff,
                                                 Aprep, bias);

  dyspn_conv_wmma<<<dim3(TILES_X, H_, B_), dim3(256), 0, stream>>>(
      guide, Aprep, bias, offs, aff);

  int npix   = B_ * H_ * W_;
  int blocks = (npix + 255) / 256;
  dyspn_iter<<<blocks, 256, 0, stream>>>(input, sp_dep, confid, offs, aff, xb0, 0);
  dyspn_iter<<<blocks, 256, 0, stream>>>(xb0,   sp_dep, confid, offs, aff, xb1, 1);
  dyspn_iter<<<blocks, 256, 0, stream>>>(xb1,   sp_dep, confid, offs, aff, out, 2);
}
